// HGT_28192165331307
// MI455X (gfx1250) — matrix operand order repaired and checked by reading, not compiled, not run on hardware
//
#include <hip/hip_runtime.h>
#include <hip/hip_bf16.h>

// ---------------------------------------------------------------------------
// HGT forward on MI455X (gfx1250), wave32 + WMMA bf16.
//  - fp32 operands pre-packed once into WMMA fragment-layout bf16 buffers
//    (halves operand bytes; packed A is L2-resident at 30.8MB << 192MB L2).
//  - GEMM: 8 waves/block, each wave owns a 16x64 tile (4 f32 accumulators,
//    4 v_wmma per 32-deep k-step, A fragment amortized 4x).
//  - B k-slabs double-buffered in LDS via async global->LDS copies
//    (global_load_async_to_lds_b128 + s_wait_asynccnt), ds_store fallback.
//  - Edge attention / segment softmax / scatter-agg: VALU + f32 atomics.
// ---------------------------------------------------------------------------

typedef __attribute__((ext_vector_type(16))) __bf16 bf16x16;
typedef __attribute__((ext_vector_type(8)))  float  f32x8;
typedef __attribute__((ext_vector_type(4)))  int    v4i;

#define NNODE 30000
#define NEDGE 480000

#if defined(__has_builtin)
#if __has_builtin(__builtin_amdgcn_global_load_async_to_lds_b128) && \
    __has_builtin(__builtin_amdgcn_s_wait_asynccnt)
#define USE_ASYNC_LDS 1
#endif
#endif

// Builtin signature (from hipcc diagnostic): param1 is int4* in AS1 (global),
// param2 the LDS-side pointer, then imm offset + imm cpol.
typedef __attribute__((address_space(1))) v4i v4i_glb;
typedef __attribute__((address_space(3))) v4i v4i_lds;

__device__ __forceinline__ __bf16 f2bf(float f) {
  unsigned u = __builtin_bit_cast(unsigned, f);
  u += 0x7FFFu + ((u >> 16) & 1u);          // round-to-nearest-even
  unsigned short h = (unsigned short)(u >> 16);
  return __builtin_bit_cast(__bf16, h);
}

__device__ __forceinline__ void copy16_to_lds(const void* g, void* l) {
#if USE_ASYNC_LDS
  __builtin_amdgcn_global_load_async_to_lds_b128(
      (v4i_glb*)(uintptr_t)g,
      (v4i_lds*)(unsigned)(uintptr_t)l,   // LDS generic addr: offset in low 32b
      0, 0);
#else
  *(uint4*)l = *(const uint4*)g;
#endif
}

__device__ __forceinline__ void wait_stage() {
#if USE_ASYNC_LDS
  __builtin_amdgcn_s_wait_asynccnt(0);
#endif
  __syncthreads();
}

__device__ __forceinline__ void atomicMaxF(float* addr, float val) {
  unsigned* ua = (unsigned*)addr;
  unsigned old = *ua;
  while (true) {
    float f = __builtin_bit_cast(float, old);
    if (f >= val) break;
    unsigned assumed = old;
    old = atomicCAS(ua, assumed, __builtin_bit_cast(unsigned, val));
    if (old == assumed) break;
  }
}

// ---------------------------------------------------------------------------
// Operand packing: fp32 (row-major with ld) -> bf16 in WMMA fragment layout.
// A fragment (16x32, 16-bit): m=lane&15, half=lane>>4,
//   VGPR v<4:  K = 8*half + 2v,  v>=4: K = 16 + 8*half + 2(v-4)
// B fragment (32x16): n=lane&15, half=lane>>4, same K mapping as A
// Packed element address: ((tile*KT + kt)*32 + lane)*16 bf16.
// ---------------------------------------------------------------------------
__global__ void pack_a_bf16(const float* __restrict__ A, int lda,
                            __bf16* __restrict__ Ap, int M, int Mt_pad, int KT) {
  int idx = blockIdx.x * blockDim.x + threadIdx.x;
  if (idx >= Mt_pad * KT * 32) return;
  int lane = idx & 31;
  int kt = (idx >> 5) % KT;
  int mt = (idx >> 5) / KT;
  int m = mt * 16 + (lane & 15);
  int half = lane >> 4;
  int k0 = kt * 32;
  __bf16* dst = Ap + (size_t)idx * 16;
  if (m >= M) {
#pragma unroll
    for (int j = 0; j < 16; ++j) dst[j] = f2bf(0.0f);
    return;
  }
  const float* Arow = A + (size_t)m * lda;
#pragma unroll
  for (int v = 0; v < 4; ++v) {
    int k = k0 + half * 8 + v * 2;
    dst[2 * v] = f2bf(Arow[k]); dst[2 * v + 1] = f2bf(Arow[k + 1]);
  }
#pragma unroll
  for (int v = 4; v < 8; ++v) {
    int k = k0 + 16 + half * 8 + (v - 4) * 2;
    dst[2 * v] = f2bf(Arow[k]); dst[2 * v + 1] = f2bf(Arow[k + 1]);
  }
}

__global__ void pack_b_bf16(const float* __restrict__ B, int ldb,
                            __bf16* __restrict__ Bp, int NT, int KT) {
  int idx = blockIdx.x * blockDim.x + threadIdx.x;
  if (idx >= NT * KT * 32) return;
  int lane = idx & 31;
  int kt = (idx >> 5) % KT;
  int half = lane >> 4;
  int n = ((idx >> 5) / KT) * 16 + (lane & 15);
  const float* Bc = B + n;
  __bf16* dst = Bp + (size_t)idx * 16;
#pragma unroll
  for (int v = 0; v < 8; ++v) {
    int k = kt * 32 + (v < 4 ? half * 8 + v * 2 : 16 + half * 8 + (v - 4) * 2);
    dst[2 * v]     = f2bf(Bc[(size_t)k * ldb]);
    dst[2 * v + 1] = f2bf(Bc[(size_t)(k + 1) * ldb]);
  }
}

// ---------------------------------------------------------------------------
// Packed GEMM: C[M,N] = Ap @ Bp (+bias). Block = 256 thr = 8 waves,
// block tile 128(M) x 64(N); wave tile 16 x 64 (4 accumulators).
// Bp k-slab (4 ntiles x 32 lanes x 32B = 4KB) staged to LDS, double-buffered.
// ---------------------------------------------------------------------------
__global__ void __launch_bounds__(256)
gemm_packed_wmma(const __bf16* __restrict__ Ap, int KTfull, int kt0,
                 const __bf16* __restrict__ Bp,
                 const float* __restrict__ bias,
                 float* __restrict__ C, int ldc, int M, int KT) {
  __shared__ __align__(32) __bf16 ldsb[2][4 * 512];
  const int t = threadIdx.x;
  const int lane = t & 31, wave = t >> 5;
  const int mt = blockIdx.x * 8 + wave;
  const int nt0 = blockIdx.y * 4;
  const int m16 = lane & 15, half = lane >> 4;
  const int cnt = t >> 6;          // 0..3 : which ntile chunk this thread stages
  const int celt = t & 63;         // 0..63: which 16B piece of that chunk

  f32x8 a0 = {}, a1 = {}, a2 = {}, a3 = {};

  // prologue stage kt=0 -> buffer 0
  copy16_to_lds(Bp + ((size_t)(nt0 + cnt) * KT + 0) * 512 + celt * 8,
                &ldsb[0][cnt * 512 + celt * 8]);
  wait_stage();

  for (int kt = 0; kt < KT; ++kt) {
    const int cur = kt & 1;
    if (kt + 1 < KT)
      copy16_to_lds(Bp + ((size_t)(nt0 + cnt) * KT + (kt + 1)) * 512 + celt * 8,
                    &ldsb[cur ^ 1][cnt * 512 + celt * 8]);

    bf16x16 af = *(const bf16x16*)(Ap + (((size_t)mt * KTfull + kt0 + kt) * 32 + lane) * 16);
    bf16x16 b0 = *(const bf16x16*)&ldsb[cur][(0 * 32 + lane) * 16];
    bf16x16 b1 = *(const bf16x16*)&ldsb[cur][(1 * 32 + lane) * 16];
    bf16x16 b2 = *(const bf16x16*)&ldsb[cur][(2 * 32 + lane) * 16];
    bf16x16 b3 = *(const bf16x16*)&ldsb[cur][(3 * 32 + lane) * 16];
    a0 = __builtin_amdgcn_wmma_f32_16x16x32_bf16(false, af, false, b0, (short)0, a0, false, false);
    a1 = __builtin_amdgcn_wmma_f32_16x16x32_bf16(false, af, false, b1, (short)0, a1, false, false);
    a2 = __builtin_amdgcn_wmma_f32_16x16x32_bf16(false, af, false, b2, (short)0, a2, false, false);
    a3 = __builtin_amdgcn_wmma_f32_16x16x32_bf16(false, af, false, b3, (short)0, a3, false, false);
    wait_stage();                  // staged buffer ready + all reads of cur done
  }

  const int row0 = mt * 16;
  f32x8 accs[4] = {a0, a1, a2, a3};
#pragma unroll
  for (int j = 0; j < 4; ++j) {
    int n = (nt0 + j) * 16 + m16;
    float bv = bias ? bias[n] : 0.0f;
#pragma unroll
    for (int v = 0; v < 8; ++v) {
      int m = row0 + v + 8 * half;   // D layout: m = v + 8*half
      if (m < M) C[(size_t)m * ldc + n] = accs[j][v] + bv;
    }
  }
}

// ---------------------------------------------------------------------------
// Elementwise / edge kernels
// ---------------------------------------------------------------------------
__global__ void fill_kernel(float* __restrict__ p, float v, long n) {
  long i = (long)blockIdx.x * blockDim.x + threadIdx.x;
  for (; i < n; i += (long)gridDim.x * blockDim.x) p[i] = v;
}

__global__ void edge_logits_max(const float* __restrict__ q, int ldq,
                                const float* __restrict__ kp,
                                const int* __restrict__ src,
                                const int* __restrict__ dst,
                                const float* __restrict__ p_rel, float rsd,
                                float* __restrict__ alpha,
                                float* __restrict__ amax,
                                int E, int H, int D) {
  int idx = blockIdx.x * blockDim.x + threadIdx.x;
  if (idx >= E * H) return;
  int e = idx / H, h = idx - e * H;
  int sn = src[e], dn = dst[e];
  const float4* __restrict__ q4 = (const float4*)(q + (size_t)dn * ldq + h * D);
  const float4* __restrict__ k4 = (const float4*)(kp + (size_t)sn * (H * D) + h * D);
  float acc = 0.0f;
  for (int d = 0; d < (D >> 2); ++d) {
    float4 a = q4[d], b = k4[d];
    acc += a.x * b.x + a.y * b.y + a.z * b.z + a.w * b.w;
  }
  acc *= p_rel[h] * rsd;
  alpha[idx] = acc;
  atomicMaxF(&amax[(size_t)dn * H + h], acc);
}

__global__ void edge_exp_sum(const int* __restrict__ dst,
                             const float* __restrict__ amax,
                             float* __restrict__ alpha,
                             float* __restrict__ denom, int E, int H) {
  int idx = blockIdx.x * blockDim.x + threadIdx.x;
  if (idx >= E * H) return;
  int e = idx / H, h = idx - e * H;
  int dn = dst[e];
  float ex = __expf(alpha[idx] - amax[(size_t)dn * H + h]);
  alpha[idx] = ex;
  atomicAdd(&denom[(size_t)dn * H + h], ex);
}

__global__ void edge_aggregate(const float* __restrict__ vp,
                               const int* __restrict__ src,
                               const int* __restrict__ dst,
                               const float* __restrict__ alpha,
                               const float* __restrict__ denom,
                               float* __restrict__ agg, int E, int H, int D) {
  int idx = blockIdx.x * blockDim.x + threadIdx.x;
  if (idx >= E * H) return;
  int e = idx / H, h = idx - e * H;
  int sn = src[e], dn = dst[e];
  float a = alpha[idx] / denom[(size_t)dn * H + h];
  const float4* __restrict__ v4 = (const float4*)(vp + (size_t)sn * (H * D) + h * D);
  float* __restrict__ ar = agg + (size_t)dn * (H * D) + h * D;
  for (int d = 0; d < (D >> 2); ++d) {
    float4 v = v4[d];
    atomicAdd(&ar[4 * d],     a * v.x);
    atomicAdd(&ar[4 * d + 1], a * v.y);
    atomicAdd(&ar[4 * d + 2], a * v.z);
    atomicAdd(&ar[4 * d + 3], a * v.w);
  }
}

__global__ void gelu_kernel(const float* __restrict__ in,
                            float* __restrict__ out, long n) {
  long i = (long)blockIdx.x * blockDim.x + threadIdx.x;
  for (; i < n; i += (long)gridDim.x * blockDim.x) {
    float x = in[i];
    out[i] = 0.5f * x * (1.0f + erff(x * 0.70710678118654752f));
  }
}

__global__ void bn_stats(const float* __restrict__ x, float* __restrict__ stats,
                         long n, int c) {
  long i = (long)blockIdx.x * blockDim.x + threadIdx.x;
  for (; i < n; i += (long)gridDim.x * blockDim.x) {
    int ch = (int)(i % c);
    float v = x[i];
    atomicAdd(&stats[ch], v);
    atomicAdd(&stats[c + ch], v * v);
  }
}

__global__ void bn_apply_relu(const float* __restrict__ x,
                              const float* __restrict__ stats,
                              const float* __restrict__ gamma,
                              const float* __restrict__ beta,
                              float* __restrict__ out, long n, int c,
                              float inv_rows) {
  long i = (long)blockIdx.x * blockDim.x + threadIdx.x;
  for (; i < n; i += (long)gridDim.x * blockDim.x) {
    int ch = (int)(i % c);
    float mean = stats[ch] * inv_rows;
    float var  = stats[c + ch] * inv_rows - mean * mean;
    float y = (x[i] - mean) * rsqrtf(var + 1e-5f) * gamma[ch] + beta[ch];
    out[i] = y > 0.0f ? y : 0.0f;
  }
}

__global__ void skip_blend(float* __restrict__ out, const float* __restrict__ h,
                           const float* __restrict__ skip, long n) {
  float s = 1.0f / (1.0f + __expf(-skip[0]));
  long i = (long)blockIdx.x * blockDim.x + threadIdx.x;
  for (; i < n; i += (long)gridDim.x * blockDim.x)
    out[i] = s * out[i] + (1.0f - s) * h[i];
}

// ---------------------------------------------------------------------------
// Orchestration
// ---------------------------------------------------------------------------
extern "C" void kernel_launch(void* const* d_in, const int* in_sizes, int n_in,
                              void* d_out, int out_size, void* d_ws, size_t ws_size,
                              hipStream_t stream) {
  const int Nn = NNODE, Ee = NEDGE;

  const float* x      = (const float*)d_in[0];
  const int*   ei     = (const int*)  d_in[1];
  const float* w_kqv1 = (const float*)d_in[2];
  const float* b_kqv1 = (const float*)d_in[3];
  const float* k_rel1 = (const float*)d_in[4];
  const float* v_rel1 = (const float*)d_in[5];
  const float* p_rel1 = (const float*)d_in[6];
  const float* w_out1 = (const float*)d_in[7];
  const float* b_out1 = (const float*)d_in[8];
  // skip1 (d_in[9]) unused: layer1 in-dim 512 != out-dim 256 -> no skip gate
  const float* bn_g   = (const float*)d_in[10];
  const float* bn_b   = (const float*)d_in[11];
  const float* w_kqv2 = (const float*)d_in[12];
  const float* b_kqv2 = (const float*)d_in[13];
  const float* k_rel2 = (const float*)d_in[14];
  const float* v_rel2 = (const float*)d_in[15];
  const float* p_rel2 = (const float*)d_in[16];
  const float* w_out2 = (const float*)d_in[17];
  const float* b_out2 = (const float*)d_in[18];
  const float* skip2  = (const float*)d_in[19];

  const int* src = ei;
  const int* dst = ei + Ee;
  float* out = (float*)d_out;

  const int Mt = (Nn + 15) / 16;            // 1875 M-tiles
  const int Mt_pad = ((Mt + 7) / 8) * 8;    // 1880, block = 8 M-tiles

  // Workspace carve-out (all slice sizes multiples of 8 floats -> 32B aligned)
  float* W = (float*)d_ws;
  size_t off = 0;
  auto alloc = [&](size_t nf) { float* p = W + off; off += nf; return p; };
  float* kqv   = alloc((size_t)Nn * 768);
  float* kp    = alloc((size_t)Nn * 256);
  float* vp    = alloc((size_t)Nn * 256);
  float* att   = alloc((size_t)Ee * 4);
  float* amax  = alloc((size_t)Nn * 4);
  float* denom = alloc((size_t)Nn * 4);
  float* agg   = alloc((size_t)Nn * 256);
  float* act   = alloc((size_t)Nn * 256);
  float* h1    = alloc((size_t)Nn * 256);
  float* tmp   = alloc((size_t)Nn * 256);
  float* stats = alloc(512);
  __bf16* Ap = (__bf16*)alloc((size_t)Mt_pad * 16 * 512 / 2); // K<=512 packed A
  __bf16* Bp = (__bf16*)alloc(262144);                        // K<=512,N<=768 packed B

  auto packA = [&](const float* A, int lda, int K) {
    int KT = K / 32; long total = (long)Mt_pad * KT * 32;
    pack_a_bf16<<<(int)((total + 255) / 256), 256, 0, stream>>>(A, lda, Ap, Nn, Mt_pad, KT);
  };
  auto packB = [&](const float* B, int ldb, int Ncols, int K) {
    int NT = Ncols / 16, KT = K / 32; long total = (long)NT * KT * 32;
    pack_b_bf16<<<(int)((total + 255) / 256), 256, 0, stream>>>(B, ldb, Bp, NT, KT);
  };
  auto gemm = [&](const float* bias, float* C, int ldc, int Ncols, int K,
                  int KTfull, int kt0) {
    dim3 grid(Mt_pad / 8, (Ncols / 16) / 4);
    gemm_packed_wmma<<<grid, 256, 0, stream>>>(Ap, KTfull, kt0, Bp, bias, C, ldc,
                                               Nn, K / 32);
  };
  auto fill = [&](float* p, float v, long n) {
    int blocks = (int)((n + 255) / 256); if (blocks > 4096) blocks = 4096;
    fill_kernel<<<blocks, 256, 0, stream>>>(p, v, n);
  };

  // ======================= Layer 1 (heads=4, D=64) =========================
  packA(x, 512, 512);
  packB(w_kqv1, 768, 768, 512);
  gemm(b_kqv1, kqv, 768, 768, 512, 16, 0);

  packA(kqv, 768, 256);                         // K-section cols [0,256)
  for (int h = 0; h < 4; ++h) {
    packB(k_rel1 + h * 4096, 64, 64, 64);
    gemm(nullptr, kp + h * 64, 256, 64, 64, 8, 2 * h);
  }
  packA(kqv + 512, 768, 256);                   // V-section cols [512,768)
  for (int h = 0; h < 4; ++h) {
    packB(v_rel1 + h * 4096, 64, 64, 64);
    gemm(nullptr, vp + h * 64, 256, 64, 64, 8, 2 * h);
  }

  fill(amax, -INFINITY, (long)Nn * 4);
  fill(denom, 1e-16f,   (long)Nn * 4);
  fill(agg, 0.0f,       (long)Nn * 256);
  {
    int blk = (Ee * 4 + 255) / 256;
    edge_logits_max<<<blk, 256, 0, stream>>>(kqv + 256, 768, kp, src, dst,
                                             p_rel1, 0.125f, att, amax, Ee, 4, 64);
    edge_exp_sum<<<blk, 256, 0, stream>>>(dst, amax, att, denom, Ee, 4);
    edge_aggregate<<<blk, 256, 0, stream>>>(vp, src, dst, att, denom, agg, Ee, 4, 64);
  }
  gelu_kernel<<<4096, 256, 0, stream>>>(agg, act, (long)Nn * 256);
  packA(act, 256, 256);
  packB(w_out1, 256, 256, 256);
  gemm(b_out1, tmp, 256, 256, 256, 8, 0);
  fill(stats, 0.0f, 512);
  bn_stats<<<4096, 256, 0, stream>>>(tmp, stats, (long)Nn * 256, 256);
  bn_apply_relu<<<4096, 256, 0, stream>>>(tmp, stats, bn_g, bn_b, h1,
                                          (long)Nn * 256, 256, 1.0f / Nn);

  // ======================= Layer 2 (heads=1, D=256) ========================
  packA(h1, 256, 256);
  packB(w_kqv2, 768, 768, 256);
  gemm(b_kqv2, kqv, 768, 768, 256, 8, 0);

  packA(kqv, 768, 256);
  packB(k_rel2, 256, 256, 256);
  gemm(nullptr, kp, 256, 256, 256, 8, 0);
  packA(kqv + 512, 768, 256);
  packB(v_rel2, 256, 256, 256);
  gemm(nullptr, vp, 256, 256, 256, 8, 0);

  fill(amax, -INFINITY, (long)Nn);
  fill(denom, 1e-16f,   (long)Nn);
  fill(agg, 0.0f,       (long)Nn * 256);
  {
    int blk = (Ee + 255) / 256;
    edge_logits_max<<<blk, 256, 0, stream>>>(kqv + 256, 768, kp, src, dst,
                                             p_rel2, 0.0625f, att, amax, Ee, 1, 256);
    edge_exp_sum<<<blk, 256, 0, stream>>>(dst, amax, att, denom, Ee, 1);
    edge_aggregate<<<blk, 256, 0, stream>>>(vp, src, dst, att, denom, agg, Ee, 1, 256);
  }
  gelu_kernel<<<4096, 256, 0, stream>>>(agg, act, (long)Nn * 256);
  packA(act, 256, 256);
  packB(w_out2, 256, 256, 256);
  gemm(b_out2, out, 256, 256, 256, 8, 0);
  skip_blend<<<4096, 256, 0, stream>>>(out, h1, skip2, (long)Nn * 256);
}
